// channel_attention_26242250178684
// MI455X (gfx1250) — compile-verified, matrix operand order checked by reading, not run on hardware
//
#include <hip/hip_runtime.h>
#include <math.h>

// ---------------------------------------------------------------------------
// Types for CDNA5 WMMA (wave32): v_wmma_f32_16x16x32_bf16
// ---------------------------------------------------------------------------
typedef __bf16 bf16_t;
typedef __attribute__((ext_vector_type(16))) __bf16 v16bf;
typedef __attribute__((ext_vector_type(8)))  __bf16 v8bf;
typedef __attribute__((ext_vector_type(4)))  float  v4f;
typedef __attribute__((ext_vector_type(8)))  float  v8f;
typedef __attribute__((ext_vector_type(4)))  unsigned int u32x4;
typedef __attribute__((ext_vector_type(8)))  int          i32x8;
typedef __attribute__((ext_vector_type(4)))  int          i32x4;

union FragU { v16bf v; v8bf h[2]; };

__device__ __forceinline__ bf16_t cvt_bf(float x)  { return (bf16_t)x; }
__device__ __forceinline__ bf16_t cvt_bf(bf16_t x) { return x; }

constexpr int LDT = 40;     // padded LDS row stride in bf16 (80 B: conflict-free)

// ---------------------------------------------------------------------------
// TDM helpers: stage a 128-row x 32-elem bf16 tile (row stride = K elems) into
// LDS, with 16 B of padding after every 64 B row (-> LDT=40 layout), zeros for
// rows beyond rows_rem. One tensor_load_to_lds per tile; tracked by TENSORcnt.
// D# packing per CDNA5 ISA 8.3/8.4.
// ---------------------------------------------------------------------------
__device__ __forceinline__ void tdm_stage_tile(const bf16_t* gaddr,
                                               unsigned lds_addr,
                                               unsigned rows_rem,
                                               unsigned row_stride_elems)
{
#if defined(__HIP_DEVICE_COMPILE__)
    const unsigned long long ga = (unsigned long long)gaddr;
    u32x4 g0 = {};
    g0[0] = 1u;                                            // count=1, user D#
    g0[1] = lds_addr;                                      // LDS byte address
    g0[2] = (unsigned)(ga & 0xffffffffu);                  // global addr lo
    g0[3] = (unsigned)((ga >> 32) & 0x01ffffffu) | (2u << 30); // hi | type=2
    i32x8 g1 = {};
    g1[0] = (1 << 16)            // data_size = 2 bytes
          | (1 << 20)            // pad_enable
          | (3 << 22)            // pad_interval: 16 DWORDs (64 B)
          | (3 << 25);           // pad_amount:    4 DWORDs (16 B)
    g1[1] = (int)(32u << 16);                // tensor_dim0[15:0] = 32
    g1[2] = (int)(rows_rem << 16);           // tensor_dim1[15:0]
    g1[3] = (int)((rows_rem >> 16) & 0xffffu) | (32 << 16); // dim1 hi | tile_dim0=32
    g1[4] = 128;                             // tile_dim1 = 128 (tile_dim2 = 0)
    g1[5] = (int)row_stride_elems;           // tensor_dim0_stride lo
    const i32x4 zz = {};
#if __clang_major__ >= 23
    const i32x8 z8 = {};
    __builtin_amdgcn_tensor_load_to_lds(g0, g1, zz, zz, z8, 0);
#else
    __builtin_amdgcn_tensor_load_to_lds(g0, g1, zz, zz, 0);
#endif
#endif
}

template <int N>
__device__ __forceinline__ void tdm_wait()
{
#if defined(__HIP_DEVICE_COMPILE__)
    __builtin_amdgcn_s_wait_tensorcnt(N);   // immediate operand required
#endif
}

// ---------------------------------------------------------------------------
// Stager<T, CK>: stages a 128x32 tile (rows x contraction-k) into LDS.
// CK = true  -> contraction stride is 1: vectorized global loads.
// CK = false -> generic strides: scalar pair loads.
// Rows are clamped to Dim-1 (no zero fill needed: clamped rows/cols only feed
// C elements that the bounds-checked epilogue never stores).
// ---------------------------------------------------------------------------
template <typename T, bool CK> struct Stager;

template <> struct Stager<float, true> {        // fp32, k-contiguous: b128 loads
    v4f  r[4];
    long goff[4];
    int  soff[4];
    __device__ void init(int tid, int base0, int Dim, long rs, long) {
        const int rA = tid >> 3;                // 0..31
        const int kA = (tid & 7) * 4;           // 0,4,...,28
#pragma unroll
        for (int i = 0; i < 4; ++i) {
            const int rr = i * 32 + rA;
            const int gr = base0 + rr;
            goff[i] = (long)(gr < Dim ? gr : (Dim - 1)) * rs + kA;
            soff[i] = rr * LDT + kA;
        }
    }
    __device__ void load(const float* p) {
#pragma unroll
        for (int i = 0; i < 4; ++i) r[i] = *(const v4f*)&p[goff[i]];
    }
    __device__ void commit(bf16_t* s) {
#pragma unroll
        for (int i = 0; i < 4; ++i) {
            union { bf16_t b[4]; unsigned long long u; } pk;
#pragma unroll
            for (int e = 0; e < 4; ++e) pk.b[e] = (bf16_t)r[i][e];
            *(unsigned long long*)&s[soff[i]] = pk.u;
        }
    }
};

template <> struct Stager<bf16_t, true> {       // bf16, k-contiguous: pure b128
    v8bf r[2];
    long goff[2];
    int  soff[2];
    __device__ void init(int tid, int base0, int Dim, long rs, long) {
        const int rA = tid >> 2;                // 0..63
        const int kA = (tid & 3) * 8;           // 0,8,16,24
#pragma unroll
        for (int i = 0; i < 2; ++i) {
            const int rr = i * 64 + rA;
            const int gr = base0 + rr;
            goff[i] = (long)(gr < Dim ? gr : (Dim - 1)) * rs + kA;
            soff[i] = rr * LDT + kA;
        }
    }
    __device__ void load(const bf16_t* p) {
#pragma unroll
        for (int i = 0; i < 2; ++i) r[i] = *(const v8bf*)&p[goff[i]];
    }
    __device__ void commit(bf16_t* s) {
#pragma unroll
        for (int i = 0; i < 2; ++i) *(v8bf*)&s[soff[i]] = r[i];
    }
};

template <typename T> struct Stager<T, false> { // generic strides: scalar pairs
    T    r0[8], r1[8];
    long goff[8];
    int  soff[8];
    long kstep;
    __device__ void init(int tid, int base0, int Dim, long rs, long ks) {
        kstep = ks;
        const int rA = tid >> 4;                // 0..15
        const int kA = (tid & 15) * 2;          // 0,2,...,30
#pragma unroll
        for (int i = 0; i < 8; ++i) {
            const int rr = i * 16 + rA;
            const int gr = base0 + rr;
            goff[i] = (long)(gr < Dim ? gr : (Dim - 1)) * rs + (long)kA * ks;
            soff[i] = rr * LDT + kA;
        }
    }
    __device__ void load(const T* p) {
#pragma unroll
        for (int i = 0; i < 8; ++i) {
            r0[i] = p[goff[i]];
            r1[i] = p[goff[i] + kstep];
        }
    }
    __device__ void commit(bf16_t* s) {
#pragma unroll
        for (int i = 0; i < 8; ++i) {
            union { bf16_t b[2]; unsigned u; } pk;
            pk.b[0] = cvt_bf(r0[i]);
            pk.b[1] = cvt_bf(r1[i]);
            *(unsigned*)&s[soff[i]] = pk.u;
        }
    }
};

// ---------------------------------------------------------------------------
// Generic strided GEMM:  C[z] = alpha * sum_b  A[z,b] (M x K)  *  B[z,b] (K x N)
//   A(m,k; b,z) = A[ z*a_zs + b*a_bs + m*a_rs + k*a_cs ]
//   B(k,n; b,z) = B[ z*b_zs + b*b_bs + k*b_rs + n*b_cs ]
//   C(m,n; z)   = C[ z*c_zs + m*c_rs + n*c_cs ]
// 256 threads = 8 waves; C tile 128x128; wave tile 32x64 (2x4 16x16 frags).
// K must be a multiple of 32 (true for every launch below).
// ---------------------------------------------------------------------------
template <typename TA, typename TB, typename TC, bool AK, bool BK>
__global__ __launch_bounds__(256)
void gemm_wmma_bf16(const TA* __restrict__ A, long a_rs, long a_cs, long a_bs, long a_zs,
                    const TB* __restrict__ B, long b_rs, long b_cs, long b_bs, long b_zs,
                    TC* __restrict__ C, long c_rs, long c_cs, long c_zs,
                    int M, int N, int K, int nbatch, float alpha)
{
    __shared__ __align__(16) bf16_t sA[128 * LDT];
    __shared__ __align__(16) bf16_t sB[128 * LDT];

    const int tid  = threadIdx.x;
    const int lane = tid & 31;
    const int wave = tid >> 5;
    const int wr   = wave & 3;
    const int wc   = wave >> 2;
    const int g    = lane >> 4;
    const int l16  = lane & 15;

    const long z = blockIdx.z;
    const TA* Az = A + z * a_zs;
    const TB* Bz = B + z * b_zs;
    TC*       Cz = C + z * c_zs;

    const int row0 = blockIdx.y * 128;
    const int col0 = blockIdx.x * 128;

    Stager<TA, AK> stA;
    Stager<TB, BK> stB;
    stA.init(tid, row0, M, a_rs, a_cs);
    stB.init(tid, col0, N, b_cs, b_rs);

    v8f acc[2][4] = {};

    const int ntk    = K >> 5;
    const int ntiles = ntk * nbatch;

    stA.load(Az);
    stB.load(Bz);

    int b = 0, k0 = 0;
    for (int t = 0; t < ntiles; ++t) {
        __syncthreads();
        stA.commit(sA);
        stB.commit(sB);
        __syncthreads();

        int nb = b, nk = k0 + 32;
        if (nk >= K) { nk = 0; ++nb; }
        if (t + 1 < ntiles) {
            stA.load(Az + (long)nb * a_bs + (long)nk * a_cs);
            stB.load(Bz + (long)nb * b_bs + (long)nk * b_rs);
        }
        b = nb; k0 = nk;

        FragU af[2], bfg[4];
#pragma unroll
        for (int i = 0; i < 2; ++i) {
            const int m = (wr * 2 + i) * 16 + l16;
            af[i].h[0] = *(const v8bf*)&sA[m * LDT + 8 * g];
            af[i].h[1] = *(const v8bf*)&sA[m * LDT + 16 + 8 * g];
        }
#pragma unroll
        for (int j = 0; j < 4; ++j) {
            const int n = (wc * 4 + j) * 16 + l16;
            bfg[j].h[0] = *(const v8bf*)&sB[n * LDT + 8 * g];
            bfg[j].h[1] = *(const v8bf*)&sB[n * LDT + 16 + 8 * g];
        }

#pragma unroll
        for (int i = 0; i < 2; ++i)
#pragma unroll
            for (int j = 0; j < 4; ++j)
                acc[i][j] = __builtin_amdgcn_wmma_f32_16x16x32_bf16(
                    false, af[i].v, false, bfg[j].v,
                    (short)0, acc[i][j], false, false);
    }

#pragma unroll
    for (int i = 0; i < 2; ++i) {
        const int mbase = row0 + (wr * 2 + i) * 16 + 8 * g;
#pragma unroll
        for (int j = 0; j < 4; ++j) {
            const int n = col0 + (wc * 4 + j) * 16 + l16;
            if (n < N) {
                long coff = (long)mbase * c_rs + (long)n * c_cs;
#pragma unroll
                for (int r = 0; r < 8; ++r, coff += c_rs) {
                    if (mbase + r < M)
                        Cz[coff] = (TC)(alpha * acc[i][j][r]);
                }
            }
        }
    }
}

// ---------------------------------------------------------------------------
// TDM-staged GEMM for bf16 x bf16 with contraction-contiguous operands:
//   A(m,k; b,z) = A[ z*a_zs + b*a_bs + m*K + k ]      (probs)
//   B(k,n; b,z) = B[ z*b_zs + b*b_bs + n*K + k ]      (V)
//   C(m,n; z)   = C[ z*c_zs + m*N + n ]               (ctx, fp32)
// Wave 0 drives the Tensor Data Mover: one tensor_load_to_lds per operand
// tile, double-buffered LDS, drained with s_wait_tensorcnt while the other
// waves run WMMAs. TDM applies the 80 B padded-row layout and zero-fills
// out-of-range rows itself.
// ---------------------------------------------------------------------------
__global__ __launch_bounds__(256)
void gemm_wmma_bf16_tdm(const bf16_t* __restrict__ A, long a_bs, long a_zs,
                        const bf16_t* __restrict__ B, long b_bs, long b_zs,
                        float* __restrict__ C, long c_zs,
                        int M, int N, int K, int nbatch, float alpha)
{
    __shared__ __align__(16) bf16_t sA[2][128 * LDT];
    __shared__ __align__(16) bf16_t sB[2][128 * LDT];

    const int tid  = threadIdx.x;
    const int lane = tid & 31;
    const int wave = tid >> 5;
    const int wr   = wave & 3;
    const int wc   = wave >> 2;
    const int g    = lane >> 4;
    const int l16  = lane & 15;

    const long z = blockIdx.z;
    const bf16_t* Az = A + z * a_zs;
    const bf16_t* Bz = B + z * b_zs;
    float*        Cz = C + z * c_zs;

    const int row0 = blockIdx.y * 128;
    const int col0 = blockIdx.x * 128;
    const unsigned rowsA = (unsigned)(M - row0);
    const unsigned rowsB = (unsigned)(N - col0);

    v8f acc[2][4] = {};

    const int ntk    = K >> 5;
    const int ntiles = ntk * nbatch;

    // tile 0 -> buffer 0
    if (wave == 0) {
        tdm_stage_tile(Az + (long)row0 * K, (unsigned)(size_t)&sA[0][0], rowsA, K);
        tdm_stage_tile(Bz + (long)col0 * K, (unsigned)(size_t)&sB[0][0], rowsB, K);
    }

    int b = 0, k0 = 0;
    for (int t = 0; t < ntiles; ++t) {
        const int cur = t & 1;

        int nb = b, nk = k0 + 32;
        if (nk >= K) { nk = 0; ++nb; }
        if (wave == 0) {
            if (t + 1 < ntiles) {
                const int nxt = cur ^ 1;
                tdm_stage_tile(Az + (long)nb * a_bs + (long)row0 * K + nk,
                               (unsigned)(size_t)&sA[nxt][0], rowsA, K);
                tdm_stage_tile(Bz + (long)nb * b_bs + (long)col0 * K + nk,
                               (unsigned)(size_t)&sB[nxt][0], rowsB, K);
                tdm_wait<2>();      // next pair may stay in flight
            } else {
                tdm_wait<0>();
            }
        }
        b = nb; k0 = nk;
        __syncthreads();            // buffer `cur` populated for all waves

        const bf16_t* bufA = sA[cur];
        const bf16_t* bufB = sB[cur];
        FragU af[2], bfg[4];
#pragma unroll
        for (int i = 0; i < 2; ++i) {
            const int m = (wr * 2 + i) * 16 + l16;
            af[i].h[0] = *(const v8bf*)&bufA[m * LDT + 8 * g];
            af[i].h[1] = *(const v8bf*)&bufA[m * LDT + 16 + 8 * g];
        }
#pragma unroll
        for (int j = 0; j < 4; ++j) {
            const int n = (wc * 4 + j) * 16 + l16;
            bfg[j].h[0] = *(const v8bf*)&bufB[n * LDT + 8 * g];
            bfg[j].h[1] = *(const v8bf*)&bufB[n * LDT + 16 + 8 * g];
        }

#pragma unroll
        for (int i = 0; i < 2; ++i)
#pragma unroll
            for (int j = 0; j < 4; ++j)
                acc[i][j] = __builtin_amdgcn_wmma_f32_16x16x32_bf16(
                    false, af[i].v, false, bfg[j].v,
                    (short)0, acc[i][j], false, false);

        __syncthreads();            // reads of `cur` done before TDM reuses it
    }

#pragma unroll
    for (int i = 0; i < 2; ++i) {
        const int mbase = row0 + (wr * 2 + i) * 16 + 8 * g;
#pragma unroll
        for (int j = 0; j < 4; ++j) {
            const int n = col0 + (wc * 4 + j) * 16 + l16;
            if (n < N) {
                long coff = (long)mbase * N + n;
#pragma unroll
                for (int r = 0; r < 8; ++r, coff += N) {
                    if (mbase + r < M)
                        Cz[coff] = alpha * acc[i][j][r];
                }
            }
        }
    }
}

// ---------------------------------------------------------------------------
// InstanceNorm2d (over [C,KV], eps=1e-5, no affine) + softmax over KV.
// ---------------------------------------------------------------------------
__global__ __launch_bounds__(256)
void norm_softmax_kernel(const float* __restrict__ scores,
                         bf16_t* __restrict__ probs, int Crows, int KV)
{
    const long z = blockIdx.x;
    const float* S = scores + z * (long)Crows * KV;
    bf16_t*      P = probs  + z * (long)Crows * KV;

    const int tid = threadIdx.x;
    const long total = (long)Crows * KV;

    float s = 0.0f, ss = 0.0f;
    for (long i = tid; i < total; i += 256) {
        const float x = S[i];
        s += x; ss += x * x;
    }
    __shared__ float red[256], red2[256];
    __shared__ float s_mu, s_rsig;
    red[tid] = s; red2[tid] = ss;
    __syncthreads();
    for (int off = 128; off > 0; off >>= 1) {
        if (tid < off) { red[tid] += red[tid + off]; red2[tid] += red2[tid + off]; }
        __syncthreads();
    }
    if (tid == 0) {
        const float inv = 1.0f / (float)total;
        const float mu  = red[0] * inv;
        const float var = red2[0] * inv - mu * mu;
        s_mu = mu; s_rsig = rsqrtf(var + 1e-5f);
    }
    __syncthreads();
    const float mu = s_mu, rsig = s_rsig;

    const int lane = tid & 31, wave = tid >> 5;
    for (int r = wave; r < Crows; r += 8) {
        const float* row = S + (long)r * KV;
        float m = -1e30f;
        for (int k = lane; k < KV; k += 32)
            m = fmaxf(m, (row[k] - mu) * rsig);
        for (int off = 16; off; off >>= 1)
            m = fmaxf(m, __shfl_xor(m, off, 32));
        float sum = 0.0f;
        for (int k = lane; k < KV; k += 32)
            sum += __expf((row[k] - mu) * rsig - m);
        for (int off = 16; off; off >>= 1)
            sum += __shfl_xor(sum, off, 32);
        const float inv = 1.0f / sum;
        bf16_t* prow = P + (long)r * KV;
        for (int k = lane; k < KV; k += 32)
            prow[k] = (bf16_t)(__expf((row[k] - mu) * rsig - m) * inv);
    }
}

// ---------------------------------------------------------------------------
// Host orchestration
// ---------------------------------------------------------------------------
extern "C" void kernel_launch(void* const* d_in, const int* in_sizes, int n_in,
                              void* d_out, int out_size, void* d_ws, size_t ws_size,
                              hipStream_t stream)
{
    constexpr int Bn = 8, Nn = 1024, KV = 960, H = 4;
    const int Cs[4] = {64, 128, 256, 512};

    const float* emb[4]  = {(const float*)d_in[0], (const float*)d_in[1],
                            (const float*)d_in[2], (const float*)d_in[3]};
    const float* emb_all = (const float*)d_in[4];
    const float* Wq[4]   = {(const float*)d_in[5], (const float*)d_in[6],
                            (const float*)d_in[7], (const float*)d_in[8]};
    const float* Wk      = (const float*)d_in[9];
    const float* Wv      = (const float*)d_in[10];
    const float* Wo[4]   = {(const float*)d_in[11], (const float*)d_in[12],
                            (const float*)d_in[13], (const float*)d_in[14]};

    char* ws = (char*)d_ws;
    const size_t szKV = (size_t)Bn * H * Nn * KV * sizeof(bf16_t);
    const size_t szQ  = (size_t)Bn * H * Nn * 512 * sizeof(bf16_t);
    const size_t szS  = (size_t)Bn * H * 512 * KV * sizeof(float);
    const size_t szP  = (size_t)Bn * H * 512 * KV * sizeof(bf16_t);

    bf16_t* Kbf = (bf16_t*)(ws);
    bf16_t* Vbf = (bf16_t*)(ws + szKV);
    bf16_t* Qbf = (bf16_t*)(ws + 2 * szKV);
    float*  Sc  = (float*) (ws + 2 * szKV + szQ);
    bf16_t* Pb  = (bf16_t*)(ws + 2 * szKV + szQ + szS);
    float*  Ctx = (float*) (ws + 2 * szKV + szQ + szS + szP);

    const dim3 blk(256);
    const float inv_sqrt_kv = 1.0f / sqrtf((float)KV);

    for (int h = 0; h < H; ++h) {
        gemm_wmma_bf16<float, float, bf16_t, true, true>
            <<<dim3((KV + 127) / 128, (Nn + 127) / 128, Bn), blk, 0, stream>>>(
                emb_all, KV, 1, 0, (long)Nn * KV,
                Wk + (size_t)h * KV * KV, 1, KV, 0, 0,
                Kbf + (size_t)h * Nn * KV, KV, 1, (long)H * Nn * KV,
                Nn, KV, KV, 1, 1.0f);
        gemm_wmma_bf16<float, float, bf16_t, true, true>
            <<<dim3((KV + 127) / 128, (Nn + 127) / 128, Bn), blk, 0, stream>>>(
                emb_all, KV, 1, 0, (long)Nn * KV,
                Wv + (size_t)h * KV * KV, 1, KV, 0, 0,
                Vbf + (size_t)h * Nn * KV, KV, 1, (long)H * Nn * KV,
                Nn, KV, KV, 1, 1.0f);
    }

    float* outp = (float*)d_out;
    for (int br = 0; br < 4; ++br) {
        const int c = Cs[br];

        for (int h = 0; h < H; ++h) {
            gemm_wmma_bf16<float, float, bf16_t, true, true>
                <<<dim3((c + 127) / 128, (Nn + 127) / 128, Bn), blk, 0, stream>>>(
                    emb[br], c, 1, 0, (long)Nn * c,
                    Wq[br] + (size_t)h * c * c, 1, c, 0, 0,
                    Qbf + (size_t)h * Nn * c, c, 1, (long)H * Nn * c,
                    Nn, c, c, 1, 1.0f);
        }

        gemm_wmma_bf16<bf16_t, bf16_t, float, false, false>
            <<<dim3((KV + 127) / 128, (c + 127) / 128, Bn * H), blk, 0, stream>>>(
                Qbf, 1, c, 0, (long)Nn * c,
                Kbf, KV, 1, 0, (long)Nn * KV,
                Sc, KV, 1, (long)c * KV,
                c, KV, Nn, 1, inv_sqrt_kv);

        norm_softmax_kernel<<<Bn * H, blk, 0, stream>>>(Sc, Pb, c, KV);

        // ctx via the Tensor Data Mover path
        gemm_wmma_bf16_tdm
            <<<dim3((Nn + 127) / 128, (c + 127) / 128, Bn), blk, 0, stream>>>(
                Pb, (long)c * KV, (long)H * c * KV,
                Vbf, (long)Nn * KV, (long)H * Nn * KV,
                Ctx, (long)c * Nn,
                c, Nn, KV, H, 1.0f / (float)H);

        gemm_wmma_bf16<float, float, float, false, true>
            <<<dim3((c + 127) / 128, (Nn + 127) / 128, Bn), blk, 0, stream>>>(
                Ctx, 1, Nn, 0, (long)c * Nn,
                Wo[br], 1, c, 0, 0,
                outp, c, 1, (long)Nn * c,
                Nn, c, c, 1, 1.0f);

        outp += (size_t)Bn * Nn * c;
    }
}
